// NovelCosClassifier_40372692583116
// MI455X (gfx1250) — compile-verified, matrix operand order checked by reading, not run on hardware
//
#include <hip/hip_runtime.h>
#include <hip/hip_bf16.h>
#include <math.h>

// ---------------- problem constants (from reference) ----------------
#define B_ROWS  2048
#define D_DIM   25088
#define C_CLS   100
#define C_PAD   112            // 7 N-tiles of 16
#define N_WAVES 7              // one wave per N-tile
#define K_SLICES 4
#define K_SLICE (D_DIM / K_SLICES)   // 6272, multiple of 32
#define N_CHUNKS (D_DIM / 32)        // 784 K-chunks total
#define EPS_N   1e-8f
#define T_SCALE 10.0f

typedef __attribute__((ext_vector_type(16))) __bf16 v16bf;
typedef __attribute__((ext_vector_type(8)))  float  v8f;

// split fp32 -> hi bf16 + residual lo bf16 (3-product scheme ~ fp32 accurate)
__device__ __forceinline__ void cvt_split(const float* v, v16bf& hi, v16bf& lo) {
#pragma unroll
  for (int i = 0; i < 16; ++i) {
    float x  = v[i];
    __bf16 h = (__bf16)x;
    hi[i] = h;
    lo[i] = (__bf16)(x - (float)h);
  }
}

// ---------------- weight row norms ----------------
__global__ __launch_bounds__(256) void wnorm_kernel(const float* __restrict__ wgt,
                                                    float* __restrict__ wn) {
  const int c = blockIdx.x;            // 0..111
  __shared__ float red[256];
  float s = 0.f;
  if (c < C_CLS) {
    const float* p = wgt + (size_t)c * D_DIM;
    for (int i = threadIdx.x; i < D_DIM; i += 256) { float x = p[i]; s += x * x; }
  }
  red[threadIdx.x] = s;
  __syncthreads();
  for (int st = 128; st > 0; st >>= 1) {
    if ((int)threadIdx.x < st) red[threadIdx.x] += red[threadIdx.x + st];
    __syncthreads();
  }
  if (threadIdx.x == 0) wn[c] = (c < C_CLS) ? fmaxf(sqrtf(red[0]), EPS_N) : 1.0f;
}

// ---------------- pre-convert weight into per-lane WMMA fragment layout ----------------
// bhi/blo layout: [ntile][chunk][lane][16] bf16; element i of lane covers
//   K = chunk*32 + (i<8 ? sel*8 + i : 16 + sel*8 + (i-8)),  sel = lane>>4
//   col = ntile*16 + (lane&15);  cols >= 100 stored as zeros.
__global__ __launch_bounds__(256) void wprep_kernel(const float* __restrict__ wgt,
                                                    __bf16* __restrict__ bhi,
                                                    __bf16* __restrict__ blo) {
  const int t = blockIdx.x * 256 + threadIdx.x;   // < 7*784*32 = 175616
  const int lane  = t & 31;
  const int cw    = t >> 5;                        // ntile*784 + chunk
  const int chunk = cw % N_CHUNKS;
  const int ntile = cw / N_CHUNKS;
  const int col   = ntile * 16 + (lane & 15);
  const int sel   = lane >> 4;
  const int o1    = chunk * 32 + sel * 8;

  union { float4 v[4]; float f[16]; } u;
  if (col < C_CLS) {
    const float* p = wgt + (size_t)col * D_DIM;
    u.v[0] = *(const float4*)(p + o1);
    u.v[1] = *(const float4*)(p + o1 + 4);
    u.v[2] = *(const float4*)(p + o1 + 16);
    u.v[3] = *(const float4*)(p + o1 + 20);
  } else {
    u.v[0] = make_float4(0.f, 0.f, 0.f, 0.f);
    u.v[1] = make_float4(0.f, 0.f, 0.f, 0.f);
    u.v[2] = make_float4(0.f, 0.f, 0.f, 0.f);
    u.v[3] = make_float4(0.f, 0.f, 0.f, 0.f);
  }
  v16bf hi, lo;
  cvt_split(u.f, hi, lo);
  const size_t off = (size_t)t * 16;               // == ((cw*32)+lane)*16
  *(v16bf*)(bhi + off) = hi;
  *(v16bf*)(blo + off) = lo;
}

// ---------------- GEMM: raw dots via bf16x3 WMMA, fused feature norms ----------------
__global__ __launch_bounds__(N_WAVES * 32) void cos_gemm_bf16x3(
    const float*  __restrict__ feat,
    const __bf16* __restrict__ bhi_p, const __bf16* __restrict__ blo_p,
    float* __restrict__ part,      // [K_SLICES][B_ROWS][C_PAD]
    float* __restrict__ fns) {     // [K_SLICES][B_ROWS] partial sum-of-squares
  const int lane  = threadIdx.x & 31;
  const int wave  = threadIdx.x >> 5;     // 0..6  -> N tile
  const int r     = lane & 15;
  const int sel   = lane >> 4;            // 0/1 (lane half)
  const int mtile = blockIdx.x;           // 0..127
  const int ks    = blockIdx.y;           // 0..3
  const int kbase = ks * K_SLICE;
  const int chunk0 = ks * (K_SLICE / 32); // 196 chunks per slice

  const int mrow = mtile * 16 + r;
  const float* aptr = feat + (size_t)mrow * D_DIM + kbase;

  // packed B fragments for this (wave, slice): advance 512 bf16 per chunk
  size_t boff = (((size_t)wave * N_CHUNKS + chunk0) * 32 + lane) * 16;

  v8f  acc   = {};
  float sumsq = 0.f;

  for (int kc = 0; kc < K_SLICE; kc += 32, boff += 512) {
    // per-lane K layout for the 16-bit A fragment:
    // lanes 0-15: K in [kc, kc+8) U [kc+16, kc+24); lanes 16-31: +8
    const int o1 = kc + sel * 8;
    union { float4 v[4]; float f[16]; } ua;
    ua.v[0] = *(const float4*)(aptr + o1);
    ua.v[1] = *(const float4*)(aptr + o1 + 4);
    ua.v[2] = *(const float4*)(aptr + o1 + 16);
    ua.v[3] = *(const float4*)(aptr + o1 + 20);

    const v16bf bhi = *(const v16bf*)(bhi_p + boff);
    const v16bf blo = *(const v16bf*)(blo_p + boff);

    // fused ||feature_row||^2 (only one wave per row-tile contributes)
    if (wave == 0) {
#pragma unroll
      for (int i = 0; i < 16; ++i) sumsq += ua.f[i] * ua.f[i];
    }

    v16bf ahi, alo;
    cvt_split(ua.f, ahi, alo);

    // (ahi+alo)*(bhi+blo) ~= ahi*bhi + ahi*blo + alo*bhi  (drop lo*lo)
    acc = __builtin_amdgcn_wmma_f32_16x16x32_bf16(false, alo, false, bhi,
                                                  (short)0, acc, false, false);
    acc = __builtin_amdgcn_wmma_f32_16x16x32_bf16(false, ahi, false, blo,
                                                  (short)0, acc, false, false);
    acc = __builtin_amdgcn_wmma_f32_16x16x32_bf16(false, ahi, false, bhi,
                                                  (short)0, acc, false, false);

    __builtin_prefetch(aptr + kc + 512, 0, 0);   // global_prefetch_b8, A stream
  }

  // store 16x16 tile: VGPR i -> M = i + sel*8, N = wave*16 + r
#pragma unroll
  for (int i = 0; i < 8; ++i) {
    const int m = mtile * 16 + i + sel * 8;
    part[((size_t)ks * B_ROWS + m) * C_PAD + wave * 16 + r] = acc[i];
  }

  if (wave == 0) {
    float tot = sumsq + __shfl_xor(sumsq, 16, 32);  // combine lane halves
    if (lane < 16) fns[(size_t)ks * B_ROWS + mtile * 16 + lane] = tot;
  }
}

// ---------------- per-row softmax CE + accuracy, deterministic reduction ----------------
__global__ __launch_bounds__(256) void loss_kernel(
    const float* __restrict__ part, const float* __restrict__ fns,
    const float* __restrict__ wn, const int* __restrict__ label,
    float* __restrict__ partials) {
  __shared__ float s_sw[C_PAD];
  __shared__ float rl[256], ra[256], rn[256];
  for (int c = threadIdx.x; c < C_PAD; c += 256) s_sw[c] = T_SCALE / wn[c];
  __syncthreads();

  const int b = blockIdx.x * 256 + threadIdx.x;    // 0..2047
  const size_t sstep = (size_t)B_ROWS * C_PAD;
  float fsq = fns[b] + fns[B_ROWS + b] + fns[2 * B_ROWS + b] + fns[3 * B_ROWS + b];
  float invf = 1.0f / fmaxf(sqrtf(fsq), EPS_N);
  const float* p0 = part + (size_t)b * C_PAD;

  const int lab = label[b];
  const bool valid = (lab >= 0);
  const int slab = valid ? lab : 0;

  float maxv = -INFINITY; int amax = 0;
  for (int c = 0; c < C_CLS; ++c) {
    float raw = p0[c] + p0[sstep + c] + p0[2 * sstep + c] + p0[3 * sstep + c];
    float l = raw * s_sw[c] * invf;
    if (l > maxv) { maxv = l; amax = c; }
  }
  float sum = 0.f, labl = 0.f;
  for (int c = 0; c < C_CLS; ++c) {
    float raw = p0[c] + p0[sstep + c] + p0[2 * sstep + c] + p0[3 * sstep + c];
    float l = raw * s_sw[c] * invf;
    sum += expf(l - maxv);
    if (c == slab) labl = l;
  }
  float nll = -(labl - maxv - logf(sum));

  rl[threadIdx.x] = valid ? nll : 0.f;
  ra[threadIdx.x] = (valid && (amax == lab)) ? 1.f : 0.f;
  rn[threadIdx.x] = valid ? 1.f : 0.f;
  __syncthreads();
  for (int st = 128; st > 0; st >>= 1) {
    if ((int)threadIdx.x < st) {
      rl[threadIdx.x] += rl[threadIdx.x + st];
      ra[threadIdx.x] += ra[threadIdx.x + st];
      rn[threadIdx.x] += rn[threadIdx.x + st];
    }
    __syncthreads();
  }
  if (threadIdx.x == 0) {
    partials[blockIdx.x * 4 + 0] = rl[0];
    partials[blockIdx.x * 4 + 1] = ra[0];
    partials[blockIdx.x * 4 + 2] = rn[0];
  }
}

__global__ void finalize_kernel(const float* __restrict__ partials,
                                float* __restrict__ out) {
  if (threadIdx.x == 0 && blockIdx.x == 0) {
    float ls = 0.f, as = 0.f, nv = 0.f;
    for (int i = 0; i < B_ROWS / 256; ++i) {   // ordered -> deterministic
      ls += partials[i * 4 + 0];
      as += partials[i * 4 + 1];
      nv += partials[i * 4 + 2];
    }
    out[0] = ls / fmaxf(nv, 1.0f);
    out[1] = as / (nv + 1e-10f);
  }
}

// ---------------- launch ----------------
extern "C" void kernel_launch(void* const* d_in, const int* in_sizes, int n_in,
                              void* d_out, int out_size, void* d_ws, size_t ws_size,
                              hipStream_t stream) {
  const float* feat  = (const float*)d_in[0];   // [2048, 25088] fp32
  const int*   label = (const int*)d_in[1];     // [2048]
  const float* wgt   = (const float*)d_in[2];   // [100, 25088] fp32
  float* out = (float*)d_out;                   // [loss, acc]

  // ---- workspace carve-up ----
  float* ws       = (float*)d_ws;
  float* part     = ws;                                          // 4*2048*112 f32
  float* fns      = part + (size_t)K_SLICES * B_ROWS * C_PAD;    // 4*2048 f32
  float* wn       = fns + K_SLICES * B_ROWS;                     // 112 f32
  float* partials = wn + C_PAD;                                  // 8*4 f32
  __bf16* bhi     = (__bf16*)(partials + 32);                    // 7*784*512 bf16
  __bf16* blo     = bhi + (size_t)N_WAVES * N_CHUNKS * 512;      // same size

  hipLaunchKernelGGL(wnorm_kernel, dim3(C_PAD), dim3(256), 0, stream, wgt, wn);
  hipLaunchKernelGGL(wprep_kernel, dim3((N_WAVES * N_CHUNKS * 32) / 256),
                     dim3(256), 0, stream, wgt, bhi, blo);
  hipLaunchKernelGGL(cos_gemm_bf16x3, dim3(B_ROWS / 16, K_SLICES),
                     dim3(N_WAVES * 32), 0, stream, feat, bhi, blo, part, fns);
  hipLaunchKernelGGL(loss_kernel, dim3(B_ROWS / 256), dim3(256), 0, stream,
                     part, fns, wn, label, partials);
  hipLaunchKernelGGL(finalize_kernel, dim3(1), dim3(1), 0, stream, partials, out);
}
